// SingleHeadAttention_32023276159091
// MI455X (gfx1250) — matrix operand order repaired and checked by reading, not compiled, not run on hardware
//
#include <hip/hip_runtime.h>
#include <stdint.h>

// ---------------------------------------------------------------------------
// Single-head causal attention, B=4 T=4096 D=1024 H=64, fp32 in/out.
// f16 WMMA (v_wmma_f32_16x16x32_f16) everywhere, fp32 softmax/accumulate,
// flash-attention streaming. Kernel A: fused QKV projection (x read once).
// Kernel B: block-cooperative K/V staging in LDS, folded-band load balance.
// ---------------------------------------------------------------------------

#define T_SEQ   4096
#define DMODEL  1024
#define HDIM    64
#define BT      16384   // B * T

typedef __attribute__((ext_vector_type(16))) _Float16 v16h;
typedef __attribute__((ext_vector_type(8)))  float    v8f;

union FragH { uint32_t u[8]; v16h v; };
union FragF { float    f[8]; v8f  v; };

static __device__ __forceinline__ uint32_t pack_f16(float a, float b) {
  union { _Float16 h[2]; uint32_t u; } x;
  x.h[0] = (_Float16)a;         // low 16 bits  = element k
  x.h[1] = (_Float16)b;         // high 16 bits = element k+1
  return x.u;
}

static __device__ __forceinline__ unsigned short f16_bits(float a) {
  union { _Float16 h; unsigned short u; } x;
  x.h = (_Float16)a;
  return x.u;
}

// ---------------------------------------------------------------------------
// Kernel A: fused QKV projection. grid = BT/128 blocks, 256 threads (8 waves).
// Each block: 128 M-rows; x tile staged once in LDS (f16), Wq/Wk/Wv chunks
// staged in WMMA-B-ready layout; each wave runs 24 WMMAs per 64-wide K chunk.
// Q pre-scaled by 1/sqrt(H); V stored transposed (Vt[b][h][t]).
// ---------------------------------------------------------------------------
__global__ __launch_bounds__(256) void qkv_proj_fused(
    const float* __restrict__ x,
    const float* __restrict__ Wq, const float* __restrict__ bq,
    const float* __restrict__ Wk, const float* __restrict__ bk,
    const float* __restrict__ Wv, const float* __restrict__ bv,
    _Float16* __restrict__ qws, _Float16* __restrict__ kws,
    _Float16* __restrict__ vtws)
{
  __shared__ uint32_t sX[128 * 32];   // 16 KB: x tile [128 rows][64 k] f16
  __shared__ uint32_t sW[3 * 2048];   // 24 KB: [proj][s][t][lane][vgpr]

  const int tid  = threadIdx.x;
  const int wave = tid >> 5;
  const int lane = tid & 31;
  const int g    = lane >> 4;
  const int n    = lane & 15;
  const int row0 = blockIdx.x * 128;

  const float* Ws[3] = {Wq, Wk, Wv};
  const float* bs[3] = {bq, bk, bv};

  FragF acc[3][4];
  #pragma unroll
  for (int p = 0; p < 3; ++p)
    #pragma unroll
    for (int t = 0; t < 4; ++t) {
      const float bb = bs[p][t * 16 + n];
      #pragma unroll
      for (int v = 0; v < 8; ++v) acc[p][t].f[v] = bb;
    }

  const float2* x2 = (const float2*)x;

  for (int kc = 0; kc < 16; ++kc) {      // 64-wide K chunks of DMODEL
    __syncthreads();
    // x tile: row-major f16, coalesced float2 reads
    #pragma unroll
    for (int i = 0; i < 16; ++i) {
      const int e = tid + i * 256;              // e < 4096
      const int r = e >> 5, ci = e & 31;
      const float2 xv = x2[(size_t)(row0 + r) * 512 + kc * 32 + ci];
      sX[e] = pack_f16(xv.x, xv.y);
    }
    // W chunks in B-fragment layout, contiguous k per lane half (matches A): k = s*32+(L/16)*16+2v
    #pragma unroll
    for (int i = 0; i < 24; ++i) {
      const int e = tid + i * 256;              // e < 6144
      const int p = e >> 11;
      const int v = e & 7, L = (e >> 3) & 31, t = (e >> 8) & 3, s = (e >> 10) & 1;
      const int k   = kc * 64 + s * 32 + (L >> 4) * 16 + 2 * v;
      const int col = t * 16 + (L & 15);
      const float* W = Ws[p];
      sW[e] = pack_f16(W[k * HDIM + col], W[(k + 1) * HDIM + col]);
    }
    __syncthreads();

    #pragma unroll
    for (int s = 0; s < 2; ++s) {
      FragH a;
      #pragma unroll
      for (int v = 0; v < 8; ++v) {
        const int kl = g * 16 + v * 2;
        a.u[v] = sX[(wave * 16 + n) * 32 + ((s * 32 + kl) >> 1)];
      }
      #pragma unroll
      for (int p = 0; p < 3; ++p)
        #pragma unroll
        for (int t = 0; t < 4; ++t) {
          FragH bf;
          #pragma unroll
          for (int v = 0; v < 8; ++v)
            bf.u[v] = sW[(((p * 2 + s) * 4 + t) * 32 + lane) * 8 + v];
          acc[p][t].v = __builtin_amdgcn_wmma_f32_16x16x32_f16(
              false, a.v, false, bf.v, (short)0, acc[p][t].v, false, false);
        }
    }
  }

  // D layout: row = v + 8g, col = t*16 + n
  const float scale = 0.125f;   // 1/sqrt(64) folded into Q
  #pragma unroll
  for (int t = 0; t < 4; ++t)
    #pragma unroll
    for (int v = 0; v < 8; ++v) {
      const int row = row0 + wave * 16 + v + 8 * g;
      const int col = t * 16 + n;
      qws[(size_t)row * HDIM + col] = (_Float16)(acc[0][t].f[v] * scale);
      kws[(size_t)row * HDIM + col] = (_Float16)(acc[1][t].f[v]);
      const int bb = row >> 12;            // / T
      const int tp = row & (T_SEQ - 1);
      vtws[((size_t)(bb * HDIM + col)) * T_SEQ + tp] = (_Float16)(acc[2][t].f[v]);
    }
}

// ---------------------------------------------------------------------------
// Kernel B: causal flash attention. grid = 64 blocks, 8 waves/block.
// Block processes folded band pair (i, 127-i) of 128 query rows each ->
// uniform work. Per 64-key chunk: K (row-major) and Vt staged cooperatively
// in LDS; each wave does 8 WMMAs for S=QK^T and 8 for O+=P*V.
// ---------------------------------------------------------------------------
__global__ __launch_bounds__(256) void flash_attn(
    const _Float16* __restrict__ qws, const _Float16* __restrict__ kws,
    const _Float16* __restrict__ vtws, float* __restrict__ out)
{
  __shared__ uint32_t sK[64 * 32];        //  8 KB: K chunk [key][h] f16
  __shared__ uint32_t sV[64 * 32];        //  8 KB: Vt chunk [h][t_local] f16
  __shared__ uint32_t sP[8 * 16 * 32];    // 16 KB: per-wave P tile 16x64 f16

  const int tid  = threadIdx.x;
  const int wave = tid >> 5;
  const int lane = tid & 31;
  const int g    = lane >> 4;
  const int n    = lane & 15;

  const uint32_t* qw  = (const uint32_t*)qws;
  const uint4*    kw4 = (const uint4*)kws;
  const uint4*    vw4 = (const uint4*)vtws;
  uint32_t*       myP  = &sP[wave * 512];
  unsigned short* myPh = (unsigned short*)myP;
  uint4* sK4 = (uint4*)sK;
  uint4* sV4 = (uint4*)sV;

  for (int rep = 0; rep < 2; ++rep) {
    const int band = rep ? (127 - (int)blockIdx.x) : (int)blockIdx.x;
    const int q0b  = band * 128;          // flat row base of band
    const int b    = q0b >> 12;           // / T
    const int qinb = q0b & (T_SEQ - 1);
    const int q0   = q0b + wave * 16;     // this wave's flat row base
    const int rowmax = qinb + wave * 16 + 15;   // last in-batch row of wave

    // Persistent Q A-fragments (head dim 64 = two K=32 steps)
    FragH aq[2];
    #pragma unroll
    for (int s = 0; s < 2; ++s)
      #pragma unroll
      for (int v = 0; v < 8; ++v) {
        const int kl = g * 16 + v * 2;
        aq[s].u[v] = qw[(size_t)(q0 + n) * 32 + s * 16 + (kl >> 1)];
      }

    FragF oacc[4];
    #pragma unroll
    for (int t = 0; t < 4; ++t)
      #pragma unroll
      for (int v = 0; v < 8; ++v) oacc[t].f[v] = 0.0f;
    float mrow[8], lrow[8];
    #pragma unroll
    for (int v = 0; v < 8; ++v) { mrow[v] = -1e30f; lrow[v] = 0.0f; }

    const int nch = (qinb + 128 + 63) >> 6;   // 64-key chunks for whole band
    for (int c = 0; c < nch; ++c) {
      __syncthreads();
      // Stage K chunk: 64 rows x 128B, coalesced b128
      #pragma unroll
      for (int i = 0; i < 2; ++i) {
        const int e = tid + i * 256;          // e < 512
        const int j = e >> 3, q4 = e & 7;
        sK4[e] = kw4[(size_t)(b * T_SEQ + c * 64 + j) * 8 + q4];
      }
      // Stage Vt chunk: 64 h-rows x 128B, coalesced b128
      #pragma unroll
      for (int i = 0; i < 2; ++i) {
        const int e = tid + i * 256;
        const int h = e >> 3, q4 = e & 7;
        sV4[e] = vw4[(size_t)(b * HDIM + h) * (T_SEQ / 8) + c * 8 + q4];
      }
      __syncthreads();

      if (c * 64 <= rowmax) {               // causal skip (barriers still hit)
        // S = Q K^T : 16 queries x 64 keys
        FragF sacc[4];
        #pragma unroll
        for (int t = 0; t < 4; ++t)
          #pragma unroll
          for (int v = 0; v < 8; ++v) sacc[t].f[v] = 0.0f;
        #pragma unroll
        for (int s = 0; s < 2; ++s)
          #pragma unroll
          for (int t = 0; t < 4; ++t) {
            FragH bk;
            #pragma unroll
            for (int v = 0; v < 8; ++v)
              bk.u[v] = sK[(t * 16 + n) * 32 + s * 16 + g * 8 + v];
            sacc[t].v = __builtin_amdgcn_wmma_f32_16x16x32_f16(
                false, aq[s].v, false, bk.v, (short)0, sacc[t].v, false, false);
          }

        // Online softmax (fp32). D layout: row = v + 8g, col = lane%16.
        #pragma unroll
        for (int v = 0; v < 8; ++v) {
          const int rown = qinb + wave * 16 + v + 8 * g;
          float sv[4];
          #pragma unroll
          for (int t = 0; t < 4; ++t) {
            const int key = c * 64 + t * 16 + n;
            sv[t] = (key <= rown) ? sacc[t].f[v] : -1e30f;
          }
          float mx = fmaxf(fmaxf(sv[0], sv[1]), fmaxf(sv[2], sv[3]));
          #pragma unroll
          for (int off = 1; off < 16; off <<= 1)
            mx = fmaxf(mx, __shfl_xor(mx, off, 32));
          const float mnew  = fmaxf(mrow[v], mx);
          const float alpha = __expf(mrow[v] - mnew);
          float pv[4], ps = 0.0f;
          #pragma unroll
          for (int t = 0; t < 4; ++t) { pv[t] = __expf(sv[t] - mnew); ps += pv[t]; }
          #pragma unroll
          for (int off = 1; off < 16; off <<= 1)
            ps += __shfl_xor(ps, off, 32);
          lrow[v] = lrow[v] * alpha + ps;
          mrow[v] = mnew;
          #pragma unroll
          for (int t = 0; t < 4; ++t) oacc[t].f[v] *= alpha;
          const int m = v + 8 * g;
          #pragma unroll
          for (int t = 0; t < 4; ++t)
            myPh[m * 64 + t * 16 + n] = f16_bits(pv[t]);
        }

        // LDS is in-order per wave; make the dependency explicit anyway.
        asm volatile("s_wait_dscnt 0" ::: "memory");

        // P as A-fragments (16x64 -> two K=32 steps)
        FragH ap[2];
        #pragma unroll
        for (int s2 = 0; s2 < 2; ++s2)
          #pragma unroll
          for (int v = 0; v < 8; ++v) {
            const int kl = g * 16 + v * 2;
            ap[s2].u[v] = myP[n * 32 + s2 * 16 + (kl >> 1)];
          }

        // O += P * V
        #pragma unroll
        for (int ht = 0; ht < 4; ++ht)
          #pragma unroll
          for (int s2 = 0; s2 < 2; ++s2) {
            FragH bvf;
            #pragma unroll
            for (int v = 0; v < 8; ++v)
              bvf.u[v] = sV[(ht * 16 + n) * 32 + s2 * 16 + g * 8 + v];
            oacc[ht].v = __builtin_amdgcn_wmma_f32_16x16x32_f16(
                false, ap[s2].v, false, bvf.v, (short)0, oacc[ht].v, false, false);
          }
      }
    }

    // Epilogue: O / l -> fp32 out [BT][64]
    #pragma unroll
    for (int t = 0; t < 4; ++t)
      #pragma unroll
      for (int v = 0; v < 8; ++v) {
        const int row = q0 + v + 8 * g;
        out[(size_t)row * HDIM + t * 16 + n] = oacc[t].f[v] / lrow[v];
      }
  }
}

// ---------------------------------------------------------------------------
extern "C" void kernel_launch(void* const* d_in, const int* in_sizes, int n_in,
                              void* d_out, int out_size, void* d_ws,
                              size_t ws_size, hipStream_t stream) {
  (void)in_sizes; (void)n_in; (void)out_size; (void)ws_size;
  const float* x  = (const float*)d_in[0];
  const float* Wq = (const float*)d_in[1];
  const float* bq = (const float*)d_in[2];
  const float* Wk = (const float*)d_in[3];
  const float* bk = (const float*)d_in[4];
  const float* Wv = (const float*)d_in[5];
  const float* bv = (const float*)d_in[6];

  _Float16* qws = (_Float16*)d_ws;                 // [BT][64] f16, pre-scaled
  _Float16* kws = qws + (size_t)BT * HDIM;         // [BT][64] f16
  _Float16* vt  = kws + (size_t)BT * HDIM;         // [B][64][T] f16 (transposed)

  qkv_proj_fused<<<dim3(BT / 128), 256, 0, stream>>>(x, Wq, bq, Wk, bk, Wv, bv,
                                                     qws, kws, vt);
  flash_attn<<<dim3(64), 256, 0, stream>>>(qws, kws, vt, (float*)d_out);
}
